// Head_4784593567799
// MI455X (gfx1250) — matrix operand order repaired and checked by reading, not compiled, not run on hardware
//
#include <hip/hip_runtime.h>
#include <math.h>

#define B_ 8
#define T_ 2048
#define C_ 1024
#define H_ 64

typedef _Float16 half_t;
typedef __attribute__((ext_vector_type(16))) _Float16 v16h;
typedef __attribute__((ext_vector_type(8)))  float    v8f;

// 16-bit A-matrix 16x32 K-index for element e (0..15) and lane half (0/1)
// per CDNA5 ISA 7.12.2: e<8 -> K=e+8*half, e>=8 -> K=16+(e-8)+8*half
__device__ __forceinline__ int kmapA(int e, int half) {
    return ((e < 8) ? e : (e + 8)) + 8 * half;
}

// ---------------------------------------------------------------------------
// Phase 0: one-time swizzle of [Wq|Wk|Wv] (fp32) into f16 B-fragment order:
//   flat = (((cs*3 + mat)*4 + t)*32 + lane)*16 + e
//   value = W_mat[cs*32 + kmapA(e, lane>>4)][t*16 + (lane&15)]
// 196608 elements = 384 KB, L2-resident for the whole projection phase.
// ---------------------------------------------------------------------------
__global__ __launch_bounds__(256) void wswz_kernel(
    const float* __restrict__ Wq, const float* __restrict__ Wk,
    const float* __restrict__ Wv, half_t* __restrict__ wswz)
{
    int idx  = blockIdx.x * 256 + threadIdx.x;   // 0 .. 196607
    int e    = idx & 15;
    int lane = (idx >> 4) & 31;
    int t    = (idx >> 9) & 3;
    int mm   = idx >> 11;                         // cs*3 + mat
    int mat  = mm % 3;
    int cs   = mm / 3;
    const float* Wm = (mat == 0) ? Wq : (mat == 1) ? Wk : Wv;
    int k   = kmapA(e, lane >> 4);
    int col = t * 16 + (lane & 15);
    wswz[idx] = (half_t)Wm[(size_t)(cs * 32 + k) * H_ + col];
}

// ---------------------------------------------------------------------------
// Phase 1: q|k|v = x @ [Wq|Wk|Wv].  LDS-free, barrier-free.
// grid: B*T/64 blocks of 256 threads; each wave owns 16 rows.
// A fragments built from global x (each element read exactly once);
// B fragments are 2x global_load_b128 from the swizzled W (L2-hot).
// q,k stored row-major f16; v stored directly in B-fragment order for P@V.
// ---------------------------------------------------------------------------
__global__ __launch_bounds__(256) void proj_kernel(
    const float* __restrict__ x, const half_t* __restrict__ wswz,
    half_t* __restrict__ qh, half_t* __restrict__ kh, half_t* __restrict__ vB)
{
    const int tid   = threadIdx.x;
    const int wid   = tid >> 5;
    const int lane  = tid & 31;
    const int halfq = lane >> 4;
    const int lmod  = lane & 15;
    const int row0  = blockIdx.x * 64;
    const float* xrow = x + (size_t)(row0 + wid * 16 + lmod) * C_;

    v8f acc[3][4] = {};

    for (int c0 = 0; c0 < C_; c0 += 32) {
        const int cs = c0 >> 5;
        // A fragment: two 32B chunks of this lane's row, cvt fp32->f16
        v16h a;
        {
            const float4* p0 = (const float4*)(xrow + c0 + 8 * halfq);
            const float4* p1 = (const float4*)(xrow + c0 + 16 + 8 * halfq);
            float4 u0 = p0[0], u1 = p0[1], u2 = p1[0], u3 = p1[1];
            a[0]=(half_t)u0.x;  a[1]=(half_t)u0.y;  a[2]=(half_t)u0.z;  a[3]=(half_t)u0.w;
            a[4]=(half_t)u1.x;  a[5]=(half_t)u1.y;  a[6]=(half_t)u1.z;  a[7]=(half_t)u1.w;
            a[8]=(half_t)u2.x;  a[9]=(half_t)u2.y;  a[10]=(half_t)u2.z; a[11]=(half_t)u2.w;
            a[12]=(half_t)u3.x; a[13]=(half_t)u3.y; a[14]=(half_t)u3.z; a[15]=(half_t)u3.w;
        }
        if (c0 + 32 < C_) __builtin_prefetch(xrow + c0 + 32, 0, 1);

#pragma unroll
        for (int mat = 0; mat < 3; ++mat) {
#pragma unroll
            for (int t = 0; t < 4; ++t) {
                v16h bw = *(const v16h*)(wswz +
                    ((((size_t)cs * 3 + mat) * 4 + t) * 32 + lane) * 16);
                acc[mat][t] = __builtin_amdgcn_wmma_f32_16x16x32_f16(
                    false, a, false, bw, (short)0, acc[mat][t], false, false);
            }
        }
    }

    // store results (D layout: m = r + 8*halfq, n = lmod)
#pragma unroll
    for (int t = 0; t < 4; ++t)
#pragma unroll
        for (int r = 0; r < 8; ++r) {
            int row = row0 + wid * 16 + r + 8 * halfq;
            int h   = t * 16 + lmod;
            qh[(size_t)row * H_ + h] = (half_t)acc[0][t][r];
            kh[(size_t)row * H_ + h] = (half_t)acc[1][t][r];
            // v in B-fragment order: u = row/32, k = row%32
            int u  = row >> 5;
            int kk = row & 31;
            size_t va = (((size_t)u * 4 + t) * 32 + ((kk & 16) + lmod)) * 16 + (kk & 15);
            vB[va] = (half_t)acc[2][t][r];
        }
}

// ---------------------------------------------------------------------------
// Phase 2: causal flash attention. grid (T/64, B), 256 threads.
// Barrier-free: each wave owns 16 query rows independently.
// K fragments: contiguous global b128 from row-major K (L2-resident).
// V fragments: contiguous global b128 from pre-swizzled vB.
// Only LDS use: per-wave P relayout (D -> A), in-order within a wave.
// ---------------------------------------------------------------------------
__global__ __launch_bounds__(256) void attn_kernel(
    const half_t* __restrict__ qh, const half_t* __restrict__ kh,
    const half_t* __restrict__ vB, float* __restrict__ out)
{
    __shared__ __align__(16) half_t Ps[4][16][72];

    const int tid   = threadIdx.x;
    const int wid   = tid >> 5;
    const int lane  = tid & 31;
    const int halfq = lane >> 4;
    const int lmod  = lane & 15;
    const int bb    = blockIdx.y;
    const int q0    = blockIdx.x * 64;
    const float scale = 0.03125f;  // C^-0.5 = 1/32

    // Q fragments (A layout), rows q0 + wid*16 .. +15, loaded once
    v16h aq[2];
    {
        const half_t* qrow = qh + (size_t)(bb * T_ + q0 + wid * 16 + lmod) * H_;
#pragma unroll
        for (int s = 0; s < 2; ++s)
#pragma unroll
            for (int e = 0; e < 16; ++e)
                aq[s][e] = qrow[s * 32 + 16 * halfq + e];
    }

    float row_max[8], row_sum[8];
#pragma unroll
    for (int r = 0; r < 8; ++r) { row_max[r] = -INFINITY; row_sum[r] = 0.0f; }
    v8f o[4] = {};

    const int ndiag = q0 / 64;
    for (int kt = 0; kt <= ndiag; ++kt) {
        if (kt < ndiag) {
            __builtin_prefetch(
                kh + (size_t)(bb * T_ + (kt + 1) * 64 + lane) * H_, 0, 1);
            __builtin_prefetch(
                vB + (((size_t)(bb * 64 + (kt + 1) * 2) * 4) * 32 + lane) * 16, 0, 1);
        }

        // ---- S = Q K^T (scaled, causal-masked), scores in D layout ----
        float sc[4][8];
#pragma unroll
        for (int nt = 0; nt < 4; ++nt) {
            v8f accs = {};
#pragma unroll
            for (int s = 0; s < 2; ++s) {
                v16h bk = *(const v16h*)(kh +
                    (size_t)(bb * T_ + kt * 64 + nt * 16 + lmod) * H_ +
                    16 * halfq + 32 * s);
                accs = __builtin_amdgcn_wmma_f32_16x16x32_f16(
                    false, aq[s], false, bk, (short)0, accs, false, false);
            }
            int key = kt * 64 + nt * 16 + lmod;
#pragma unroll
            for (int r = 0; r < 8; ++r) {
                int m = q0 + wid * 16 + r + 8 * halfq;
                sc[nt][r] = (key > m) ? -1.0e30f : accs[r] * scale;
            }
        }

        // ---- online softmax (row reductions in 16-lane groups) ----
#pragma unroll
        for (int r = 0; r < 8; ++r) {
            float tm = fmaxf(fmaxf(sc[0][r], sc[1][r]),
                             fmaxf(sc[2][r], sc[3][r]));
#pragma unroll
            for (int off = 1; off < 16; off <<= 1)
                tm = fmaxf(tm, __shfl_xor(tm, off, 16));
            float nm   = fmaxf(row_max[r], tm);
            float corr = __expf(row_max[r] - nm);
            row_max[r] = nm;

            float ps   = 0.0f;
            int   mloc = r + 8 * halfq;
#pragma unroll
            for (int nt = 0; nt < 4; ++nt) {
                float p = __expf(sc[nt][r] - nm);
                ps += p;
                Ps[wid][mloc][nt * 16 + lmod] = (half_t)p;
            }
#pragma unroll
            for (int off = 1; off < 16; off <<= 1)
                ps += __shfl_xor(ps, off, 16);
            row_sum[r] = row_sum[r] * corr + ps;
#pragma unroll
            for (int t = 0; t < 4; ++t) o[t][r] *= corr;
        }

        // ---- O += P V (LDS DS ops are in-order within a wave) ----
#pragma unroll
        for (int s = 0; s < 2; ++s) {
            v16h ap;
#pragma unroll
            for (int e = 0; e < 16; ++e)
                ap[e] = Ps[wid][lmod][s * 32 + 16 * halfq + e];
#pragma unroll
            for (int t = 0; t < 4; ++t) {
                v16h bv = *(const v16h*)(vB +
                    (((size_t)(bb * 64 + kt * 2 + s) * 4 + t) * 32 + lane) * 16);
                o[t] = __builtin_amdgcn_wmma_f32_16x16x32_f16(
                    false, ap, false, bv, (short)0, o[t], false, false);
            }
        }
    }

    // ---- normalize and store fp32 output ----
#pragma unroll
    for (int r = 0; r < 8; ++r) {
        float inv = 1.0f / row_sum[r];
        int m = q0 + wid * 16 + r + 8 * halfq;
#pragma unroll
        for (int t = 0; t < 4; ++t)
            out[(size_t)(bb * T_ + m) * H_ + t * 16 + lmod] = o[t][r] * inv;
    }
}

// ---------------------------------------------------------------------------
extern "C" void kernel_launch(void* const* d_in, const int* in_sizes, int n_in,
                              void* d_out, int out_size, void* d_ws, size_t ws_size,
                              hipStream_t stream) {
    (void)in_sizes; (void)n_in; (void)out_size; (void)ws_size;
    const float* x  = (const float*)d_in[0];
    const float* Wq = (const float*)d_in[1];
    const float* Wk = (const float*)d_in[2];
    const float* Wv = (const float*)d_in[3];
    float* out = (float*)d_out;

    const size_t BTH = (size_t)B_ * T_ * H_;          // 1,048,576 elements
    half_t* qh   = (half_t*)d_ws;                     // 2 MB
    half_t* kh   = qh + BTH;                          // 2 MB
    half_t* vB   = kh + BTH;                          // 2 MB (B-fragment order)
    half_t* wswz = vB + BTH;                          // 384 KB swizzled W

    wswz_kernel<<<dim3(768), dim3(256), 0, stream>>>(Wq, Wk, Wv, wswz);
    proj_kernel<<<dim3(B_ * T_ / 64), dim3(256), 0, stream>>>(
        x, wswz, qh, kh, vB);
    attn_kernel<<<dim3(T_ / 64, B_), dim3(256), 0, stream>>>(
        qh, kh, vB, out);
}
